// Raindrop_51393578664518
// MI455X (gfx1250) — compile-verified, matrix operand order checked by reading, not run on hardware
//
#include <hip/hip_runtime.h>
#include <hip/hip_bf16.h>
#include <stdint.h>

// ---------------- model dims ----------------
#define S_     64
#define E_     4
#define T_     256
#define B_     128
#define DPE    16
#define HEADS  4
#define DH     68          // 272 / 4
#define D_     272         // transformer d_model
#define DPAD   288         // D_ padded to multiple of 32 (WMMA K)
#define DFF    512
#define DFEAT  1024        // T*E
#define D3     816         // 3*D_
#define DSEMB  256
#define DFIN   528         // D_ + DSEMB
#define DFINP  544         // padded to 32
#define DSTAT  6

typedef uint16_t u16;
typedef __attribute__((ext_vector_type(16))) __bf16 v16bf;
typedef __attribute__((ext_vector_type(8)))  float  v8f;

union FragAB { v16bf v; uint4 q[2]; u16 u[16]; };

__device__ __forceinline__ float bf2f(u16 h) {
  union { uint32_t i; float f; } c; c.i = ((uint32_t)h) << 16; return c.f;
}
__device__ __forceinline__ u16 f2bf(float f) {
  union { float f; uint32_t i; } c; c.f = f;
  uint32_t x = c.i;
  x += 0x7FFFu + ((x >> 16) & 1u);         // round-to-nearest-even
  return (u16)(x >> 16);
}

// ---------------------------------------------------------------------------
// Weight repack: fp32 row-major (Ks x Ns) -> bf16 WMMA B-fragment layout,
// zero-padded to (Kp x Np).  Fragment layout (16x16x32 bf16 B matrix):
//   lane = (n%16) + 16*half ; element e<8  -> k = ktile*32 + half*8 + e
//                             element e>=8 -> k = ktile*32 + 16 + half*8 + (e-8)
// dst[((ntile*Ktiles + ktile)*32 + lane)*16 + e]
// ---------------------------------------------------------------------------
__global__ void pack_w(const float* __restrict__ src, u16* __restrict__ dst,
                       int Ks, int Ns, int Kp, int Np) {
  int idx = blockIdx.x * blockDim.x + threadIdx.x;
  int total = Kp * Np;
  if (idx >= total) return;
  int k = idx / Np, n = idx % Np;
  float val = (k < Ks && n < Ns) ? src[(size_t)k * Ns + n] : 0.0f;
  int kk = k & 31, ktile = k >> 5, ntile = n >> 4;
  int half, e;
  if (kk < 16) { half = kk >> 3; e = kk & 7; }
  else         { half = (kk - 16) >> 3; e = 8 + ((kk - 16) & 7); }
  int lane   = (n & 15) + (half << 4);
  int Ktiles = Kp >> 5;
  dst[(((size_t)ntile * Ktiles + ktile) * 32 + lane) * 16 + e] = f2bf(val);
}

// ---------------------------------------------------------------------------
// Batched GEMM:  C[b] = act(A[b] (MxKp, bf16 rowmajor) @ Wp (packed bf16) + bias)
// One wave computes a 32(M) x 64(N) strip: 2 A fragments x 4 B fragments ->
// 8 WMMAs per k-step.  STATIC software pipeline: two named fragment sets,
// k-loop unrolled by 2 (odd tail peeled) so all fragment indices are
// compile-time constants and live entirely in VGPRs (no scratch demotion).
// Block = 128 threads = 4 waves -> 32M x 256N per block.
// Tile indices are clamped (wave-uniform) so EXEC is all-ones around WMMA.
// ---------------------------------------------------------------------------
template<bool RELU, bool OUTF, bool OUTB>
__global__ void __launch_bounds__(128)
gemm_bf16(const u16* __restrict__ A, const u16* __restrict__ Wp,
          const float* __restrict__ bias,
          float* __restrict__ Cf, u16* __restrict__ Cb,
          int Kp, int Np, int Nsrc, int ldcF, int ldcB,
          size_t strideA, size_t strideCf, size_t strideCb) {
  const int lane   = threadIdx.x & 31;
  const int wave   = threadIdx.x >> 5;
  const int Ntiles = Np >> 4;
  const int Ktiles = Kp >> 5;
  const int ntbase = (blockIdx.x * 4 + wave) * 4;
  if (ntbase >= Ntiles) return;                 // wave-uniform exit
  const int half  = lane >> 4;
  const int nlane = lane & 15;
  const int mbase = blockIdx.y << 5;            // 32 rows per block-row
  const size_t batch = blockIdx.z;

  int nt[4];
#pragma unroll
  for (int i = 0; i < 4; ++i) {
    int t0 = ntbase + i;
    nt[i] = (t0 < Ntiles) ? t0 : (Ntiles - 1);  // clamp: keep EXEC uniform
  }

  v8f zero8 = {0.f, 0.f, 0.f, 0.f, 0.f, 0.f, 0.f, 0.f};
  v8f acc[2][4] = {{zero8, zero8, zero8, zero8}, {zero8, zero8, zero8, zero8}};

  const u16* arow0 = A + batch * strideA + (size_t)(mbase + nlane) * Kp + half * 8;
  const u16* arow1 = arow0 + (size_t)16 * Kp;

  FragAB fa0[2], fb0[4], fa1[2], fb1[4];

  auto loadA = [&](FragAB (&fa)[2], int kt) {
    const u16* a0 = arow0 + (size_t)kt * 32;
    const u16* a1 = arow1 + (size_t)kt * 32;
    fa[0].q[0] = *(const uint4*)(a0);
    fa[0].q[1] = *(const uint4*)(a0 + 16);
    fa[1].q[0] = *(const uint4*)(a1);
    fa[1].q[1] = *(const uint4*)(a1 + 16);
  };
  auto loadB = [&](FragAB (&fb)[4], int kt) {
#pragma unroll
    for (int i = 0; i < 4; ++i)
      fb[i].v = *(const v16bf*)(Wp + (((size_t)nt[i] * Ktiles + kt) * 32 + lane) * 16);
  };
  auto domma = [&](FragAB (&fa)[2], FragAB (&fb)[4]) {
#pragma unroll
    for (int i = 0; i < 4; ++i) {
      acc[0][i] = __builtin_amdgcn_wmma_f32_16x16x32_bf16(
                      false, fa[0].v, false, fb[i].v, (short)0, acc[0][i], false, false);
      acc[1][i] = __builtin_amdgcn_wmma_f32_16x16x32_bf16(
                      false, fa[1].v, false, fb[i].v, (short)0, acc[1][i], false, false);
    }
  };

  loadA(fa0, 0); loadB(fb0, 0);
  int kt = 0;
  for (; kt + 2 <= Ktiles; kt += 2) {
    loadA(fa1, kt + 1); loadB(fb1, kt + 1);
    if (kt + 8 < Ktiles) {                      // speculative A-stream prefetch
      __builtin_prefetch(arow0 + (size_t)(kt + 8) * 32, 0, 1);
      __builtin_prefetch(arow1 + (size_t)(kt + 8) * 32, 0, 1);
    }
    domma(fa0, fb0);                            // overlap with fa1/fb1 loads
    if (kt + 2 < Ktiles) { loadA(fa0, kt + 2); loadB(fb0, kt + 2); }
    domma(fa1, fb1);                            // overlap with fa0/fb0 loads
  }
  if (kt < Ktiles) domma(fa0, fb0);             // odd-Ktiles tail

#pragma unroll
  for (int sub = 0; sub < 2; ++sub) {
#pragma unroll
    for (int i = 0; i < 4; ++i) {
      if (ntbase + i >= Ntiles) break;
      int n = (nt[i] << 4) + nlane;
      if (n >= Nsrc) continue;
      float bv = bias[n];
#pragma unroll
      for (int r = 0; r < 8; ++r) {
        int m = mbase + (sub << 4) + (half << 3) + r;
        float v = acc[sub][i][r] + bv;
        if (RELU) v = fmaxf(v, 0.0f);
        if (OUTF) Cf[batch * strideCf + (size_t)m * ldcF + n] = v;
        if (OUTB) Cb[batch * strideCb + (size_t)m * ldcB + n] = f2bf(v);
      }
    }
  }
}

// ---------------------------------------------------------------------------
// hn[b][s][t*E+e] = relu(src[t][b][s] * R_u[s*E+e])   (bf16)
// ---------------------------------------------------------------------------
__global__ void prep_hn(const float* __restrict__ src, const float* __restrict__ Ru,
                        u16* __restrict__ hn) {
  size_t idx = (size_t)blockIdx.x * blockDim.x + threadIdx.x;
  size_t total = (size_t)B_ * S_ * DFEAT;
  if (idx >= total) return;
  int f = (int)(idx % DFEAT);
  int s = (int)((idx / DFEAT) % S_);
  int b = (int)(idx / ((size_t)DFEAT * S_));
  int t = f >> 2, e = f & 3;
  float x = src[((size_t)t * B_ + b) * (2 * S_) + s];
  float h = fmaxf(x * Ru[s * E_ + e], 0.0f);
  hn[idx] = f2bf(h);
}

// ---------------------------------------------------------------------------
// sc[b][i][j] = (k_i . q_j) / 32   (LDS-tiled, bf16 in / fp32 accum)
// ---------------------------------------------------------------------------
__global__ void __launch_bounds__(256)
graph_scores(const u16* __restrict__ K, const u16* __restrict__ Q,
             float* __restrict__ sc) {
  int b = blockIdx.x;
  const u16* Kb = K + (size_t)b * S_ * DFEAT;
  const u16* Qb = Q + (size_t)b * S_ * DFEAT;
  __shared__ u16 sk[S_][68];
  __shared__ u16 sq[S_][68];
  float acc[16];
#pragma unroll
  for (int p = 0; p < 16; ++p) acc[p] = 0.f;
  for (int kc = 0; kc < DFEAT; kc += 64) {
    __syncthreads();
    for (int t = threadIdx.x; t < S_ * 64; t += 256) {
      int r = t >> 6, c = t & 63;
      sk[r][c] = Kb[(size_t)r * DFEAT + kc + c];
      sq[r][c] = Qb[(size_t)r * DFEAT + kc + c];
    }
    __syncthreads();
#pragma unroll
    for (int p = 0; p < 16; ++p) {
      int pair = threadIdx.x + 256 * p;
      int i = pair >> 6, j = pair & 63;
      float a = acc[p];
      for (int c = 0; c < 64; ++c) a += bf2f(sk[i][c]) * bf2f(sq[j][c]);
      acc[p] = a;
    }
  }
#pragma unroll
  for (int p = 0; p < 16; ++p)
    sc[(size_t)b * S_ * S_ + threadIdx.x + 256 * p] = acc[p] * 0.03125f;
}

// ---------------------------------------------------------------------------
// alpha[b][:,j] = softmax_i( sc[b][i][j] * ew[b][i][j] )     (ew==null -> 1)
// ---------------------------------------------------------------------------
__global__ void __launch_bounds__(64)
graph_alpha(const float* __restrict__ sc, const float* __restrict__ ew,
            float* __restrict__ alpha) {
  int j = blockIdx.x, b = blockIdx.y, i = threadIdx.x;
  size_t off = ((size_t)b * S_ + i) * S_ + j;
  float w = ew ? ew[off] : 1.0f;
  float s = sc[off] * w;
  __shared__ float red[64];
  red[i] = s; __syncthreads();
  for (int o = 32; o > 0; o >>= 1) { if (i < o) red[i] = fmaxf(red[i], red[i + o]); __syncthreads(); }
  float mx = red[0]; __syncthreads();
  float e = __expf(s - mx);
  red[i] = e; __syncthreads();
  for (int o = 32; o > 0; o >>= 1) { if (i < o) red[i] += red[i + o]; __syncthreads(); }
  alpha[off] = e / red[0];
}

// ---------------------------------------------------------------------------
// out[b][j][f] = sum_i alpha[b][i][j]*v[b][i][f] + skip[b][j][f]
// TO_XT: scatter into xt[t][b][j*4+e]  (f = t*4+e), else bf16 row-major.
// ---------------------------------------------------------------------------
template<bool TO_XT>
__global__ void __launch_bounds__(256)
graph_out(const float* __restrict__ alpha, const u16* __restrict__ V,
          const u16* __restrict__ Sk, u16* __restrict__ outb,
          float* __restrict__ xt) {
  int j = blockIdx.x, b = blockIdx.y;
  __shared__ float a[S_];
  if (threadIdx.x < S_) a[threadIdx.x] = alpha[((size_t)b * S_ + threadIdx.x) * S_ + j];
  __syncthreads();
  const u16* Vb  = V  + (size_t)b * S_ * DFEAT;
  const u16* Skb = Sk + ((size_t)b * S_ + j) * DFEAT;
  for (int f = threadIdx.x; f < DFEAT; f += 256) {
    float acc = bf2f(Skb[f]);
    for (int i = 0; i < S_; ++i) acc += a[i] * bf2f(Vb[(size_t)i * DFEAT + f]);
    if (TO_XT) {
      int t = f >> 2, e = f & 3;
      xt[((size_t)t * B_ + b) * D_ + j * 4 + e] = acc;
    } else {
      outb[((size_t)b * S_ + j) * DFEAT + f] = f2bf(acc);
    }
  }
}

// ---------------------------------------------------------------------------
// positional encoding into xt cols [256, 272)
// ---------------------------------------------------------------------------
__global__ void pe_kernel(const float* __restrict__ times, float* __restrict__ xt) {
  int idx = blockIdx.x * blockDim.x + threadIdx.x;   // t*B + b
  if (idx >= T_ * B_) return;
  float tm = times[idx];
  float* row = xt + (size_t)idx * D_ + S_ * E_;
#pragma unroll
  for (int p = 0; p < 8; ++p) {
    float ts = __powf(256.0f, (float)p * (1.0f / 7.0f));
    float sv = tm / ts;
    row[p]     = __sinf(sv);
    row[p + 8] = __cosf(sv);
  }
}

__global__ void static_emb_kernel(const float* __restrict__ st, const float* __restrict__ W,
                                  const float* __restrict__ bv, float* __restrict__ semb) {
  int idx = blockIdx.x * blockDim.x + threadIdx.x;
  if (idx >= B_ * DSEMB) return;
  int j = idx % DSEMB, b = idx / DSEMB;
  float a = bv[j];
#pragma unroll
  for (int k = 0; k < DSTAT; ++k) a += st[b * DSTAT + k] * W[k * DSEMB + j];
  semb[idx] = a;
}

__global__ void f32_to_bf16_pad(const float* __restrict__ src, u16* __restrict__ dst,
                                int rows, int Cs, int Cp) {
  size_t idx = (size_t)blockIdx.x * blockDim.x + threadIdx.x;
  size_t total = (size_t)rows * Cp;
  if (idx >= total) return;
  int c = (int)(idx % Cp);
  size_t r = idx / Cp;
  dst[idx] = (c < Cs) ? f2bf(src[r * Cs + c]) : (u16)0;
}

__global__ void zero_u16(u16* __restrict__ p, size_t n) {
  size_t i = (size_t)blockIdx.x * blockDim.x + threadIdx.x;
  if (i < n) p[i] = 0;
}

// ---------------------------------------------------------------------------
// attention: per (t,b,h): sc_s = q_t.k_s/sqrt(68), key-pad mask, softmax,
// ao = sum_s att_s * v_s.   qkv bf16 (T,B,816): q|k|v at 0/272/544.
// ---------------------------------------------------------------------------
__global__ void __launch_bounds__(256)
attention(const u16* __restrict__ qkv, const int* __restrict__ lengths,
          u16* __restrict__ ao) {
  int t = blockIdx.x, b = blockIdx.y, h = blockIdx.z;
  __shared__ float sq[DH];
  __shared__ float att[T_];
  __shared__ float red[256];
  const u16* qrow = qkv + ((size_t)t * B_ + b) * D3 + h * DH;
  if (threadIdx.x < DH) sq[threadIdx.x] = bf2f(qrow[threadIdx.x]);
  __syncthreads();
  int s = threadIdx.x;
  int len = lengths[b];
  const u16* krow = qkv + ((size_t)s * B_ + b) * D3 + D_ + h * DH;
  float sc = 0.f;
  for (int d = 0; d < DH; ++d) sc += sq[d] * bf2f(krow[d]);
  sc *= 0.121267813f;                          // 1/sqrt(68)
  if (s >= len) sc = -1e9f;
  red[s] = sc; __syncthreads();
  for (int o = 128; o > 0; o >>= 1) { if (s < o) red[s] = fmaxf(red[s], red[s + o]); __syncthreads(); }
  float mx = red[0]; __syncthreads();
  float e = __expf(sc - mx);
  red[s] = e; __syncthreads();
  for (int o = 128; o > 0; o >>= 1) { if (s < o) red[s] += red[s + o]; __syncthreads(); }
  att[s] = e / red[0];
  __syncthreads();
  for (int d = threadIdx.x; d < DH; d += 256) {
    float acc = 0.f;
    for (int ss = 0; ss < T_; ++ss)
      acc += att[ss] * bf2f(qkv[((size_t)ss * B_ + b) * D3 + 2 * D_ + h * DH + d]);
    ao[((size_t)t * B_ + b) * DPAD + h * DH + d] = f2bf(acc);
  }
}

// ---------------------------------------------------------------------------
// xt = LN(xt + y) ; also writes padded bf16 copy (row stride DPAD, pad zeros)
// ---------------------------------------------------------------------------
__global__ void __launch_bounds__(256)
residual_ln(const float* __restrict__ x, const float* __restrict__ y,
            const float* __restrict__ g, const float* __restrict__ bb,
            float* __restrict__ xout, u16* __restrict__ xbp) {
  int row = blockIdx.x, tid = threadIdx.x;
  const float* xr = x + (size_t)row * D_;
  const float* yr = y + (size_t)row * D_;
  float v0 = (tid < D_)       ? xr[tid] + yr[tid]             : 0.f;
  float v1 = (tid + 256 < D_) ? xr[tid + 256] + yr[tid + 256] : 0.f;
  __shared__ float red[256];
  red[tid] = v0 + v1; __syncthreads();
  for (int o = 128; o > 0; o >>= 1) { if (tid < o) red[tid] += red[tid + o]; __syncthreads(); }
  float mean = red[0] * (1.0f / D_);
  __syncthreads();
  float d0 = (tid < D_) ? v0 - mean : 0.f;
  float d1 = (tid + 256 < D_) ? v1 - mean : 0.f;
  red[tid] = d0 * d0 + d1 * d1; __syncthreads();
  for (int o = 128; o > 0; o >>= 1) { if (tid < o) red[tid] += red[tid + o]; __syncthreads(); }
  float rstd = rsqrtf(red[0] * (1.0f / D_) + 1e-5f);
  float* xo = xout + (size_t)row * D_;
  u16*   xb = xbp  + (size_t)row * DPAD;
  if (tid < D_)       { float o = d0 * rstd * g[tid] + bb[tid];             xo[tid] = o;       xb[tid] = f2bf(o); }
  if (tid + 256 < D_) { float o = d1 * rstd * g[tid + 256] + bb[tid + 256]; xo[tid + 256] = o; xb[tid + 256] = f2bf(o); }
  if (tid < DPAD - D_) xb[D_ + tid] = 0;
}

// ---------------------------------------------------------------------------
// masked mean-pool over t (t < len), / (len+1); concat static_emb; bf16 pad 544
// ---------------------------------------------------------------------------
__global__ void __launch_bounds__(256)
pool_feat(const float* __restrict__ xt, const int* __restrict__ lengths,
          const float* __restrict__ semb, u16* __restrict__ featb) {
  int b = blockIdx.x;
  int len = lengths[b];
  float inv = 1.0f / ((float)len + 1.0f);
  for (int d = threadIdx.x; d < DFINP; d += 256) {
    float v;
    if (d < D_) {
      float s = 0.f;
      for (int t = 0; t < len; ++t) s += xt[((size_t)t * B_ + b) * D_ + d];
      v = s * inv;
    } else if (d < DFIN) {
      v = semb[(size_t)b * DSEMB + (d - D_)];
    } else v = 0.f;
    featb[(size_t)b * DFINP + d] = f2bf(v);
  }
}

// ===========================================================================
extern "C" void kernel_launch(void* const* d_in, const int* in_sizes, int n_in,
                              void* d_out, int out_size, void* d_ws, size_t ws_size,
                              hipStream_t stream) {
  (void)in_sizes; (void)n_in; (void)out_size; (void)ws_size;
  const float* src      = (const float*)d_in[0];
  const float* times    = (const float*)d_in[1];
  const int*   lengths  = (const int*)  d_in[2];
  const float* statics  = (const float*)d_in[3];
  const float* R_u      = (const float*)d_in[4];
  const float* static_W = (const float*)d_in[5];
  const float* static_b = (const float*)d_in[6];
  const float* pW[2][4] = {{(const float*)d_in[7],(const float*)d_in[8],(const float*)d_in[9],(const float*)d_in[10]},
                           {(const float*)d_in[15],(const float*)d_in[16],(const float*)d_in[17],(const float*)d_in[18]}};
  const float* pB[2][4] = {{(const float*)d_in[11],(const float*)d_in[12],(const float*)d_in[13],(const float*)d_in[14]},
                           {(const float*)d_in[19],(const float*)d_in[20],(const float*)d_in[21],(const float*)d_in[22]}};
  const float* t_Wqkv = (const float*)d_in[23]; const float* t_bqkv = (const float*)d_in[24];
  const float* t_Wo   = (const float*)d_in[25]; const float* t_bo   = (const float*)d_in[26];
  const float* t_W1   = (const float*)d_in[27]; const float* t_b1   = (const float*)d_in[28];
  const float* t_W2   = (const float*)d_in[29]; const float* t_b2   = (const float*)d_in[30];
  const float* ln1g   = (const float*)d_in[31]; const float* ln1b   = (const float*)d_in[32];
  const float* ln2g   = (const float*)d_in[33]; const float* ln2b   = (const float*)d_in[34];
  const float* mlp_W1 = (const float*)d_in[35]; const float* mlp_b1 = (const float*)d_in[36];
  const float* mlp_W2 = (const float*)d_in[37]; const float* mlp_b2 = (const float*)d_in[38];
  float* out = (float*)d_out;

  // ---- workspace carve ----
  char* wsp = (char*)d_ws;
  auto alloc = [&](size_t bytes) { char* p = wsp; wsp += (bytes + 255) & ~(size_t)255; return p; };
  u16*   hn   = (u16*)alloc((size_t)B_ * S_ * DFEAT * 2);
  u16*   pwp[2][4];
  for (int l = 0; l < 2; ++l) for (int w = 0; w < 4; ++w) pwp[l][w] = (u16*)alloc((size_t)DFEAT * DFEAT * 2);
  // arena: graph q/k/v/skip (4 x 16.8MB) reused later for transformer qkv (53.5MB)
  size_t gstride = (size_t)B_ * S_ * DFEAT;                        // elems
  char*  arena = alloc(((size_t)T_ * B_ * D3 * 2 > 4 * gstride * 2) ? (size_t)T_ * B_ * D3 * 2 : 4 * gstride * 2);
  u16* gq = (u16*)arena; u16* gk = gq + gstride; u16* gv = gk + gstride; u16* gs = gv + gstride;
  u16* tqkv = (u16*)arena;
  float* sc  = (float*)alloc((size_t)B_ * S_ * S_ * 4);
  float* a1  = (float*)alloc((size_t)B_ * S_ * S_ * 4);
  float* a2  = (float*)alloc((size_t)B_ * S_ * S_ * 4);
  u16*   h1  = (u16*)alloc((size_t)B_ * S_ * DFEAT * 2);
  float* xt  = (float*)alloc((size_t)T_ * B_ * D_ * 4);
  u16*   xtb = (u16*)alloc((size_t)T_ * B_ * DPAD * 2);
  u16* Wqkvp[2], *Wop[2], *W1p[2], *W2p[2];
  for (int l = 0; l < 2; ++l) {
    Wqkvp[l] = (u16*)alloc((size_t)DPAD * D3 * 2);
    Wop[l]   = (u16*)alloc((size_t)DPAD * DPAD * 2);
    W1p[l]   = (u16*)alloc((size_t)DPAD * DFF * 2);
    W2p[l]   = (u16*)alloc((size_t)DFF * DPAD * 2);
  }
  u16*   ao   = (u16*)alloc((size_t)T_ * B_ * DPAD * 2);
  float* tmp  = (float*)alloc((size_t)T_ * B_ * D_ * 4);
  u16*   ff1  = (u16*)alloc((size_t)T_ * B_ * DFF * 2);
  float* semb = (float*)alloc((size_t)B_ * DSEMB * 4);
  u16*   featb  = (u16*)alloc((size_t)B_ * DFINP * 2);
  u16*   hidden = (u16*)alloc((size_t)B_ * DFINP * 2);
  u16*   mW1p   = (u16*)alloc((size_t)DFINP * DFIN * 2);
  u16*   mW2p   = (u16*)alloc((size_t)DFINP * 16 * 2);

  auto packs = [&](const float* s, u16* d, int Ks, int Ns, int Kp, int Np) {
    int tot = Kp * Np;
    pack_w<<<(tot + 255) / 256, 256, 0, stream>>>(s, d, Ks, Ns, Kp, Np);
  };

  // ---- prep + weight packing ----
  {
    size_t tot = (size_t)B_ * S_ * DFEAT;
    prep_hn<<<(unsigned)((tot + 255) / 256), 256, 0, stream>>>(src, R_u, hn);
  }
  for (int l = 0; l < 2; ++l)
    for (int w = 0; w < 4; ++w)
      packs(pW[l][w], pwp[l][w], DFEAT, DFEAT, DFEAT, DFEAT);
  for (int l = 0; l < 2; ++l) {
    packs(t_Wqkv + (size_t)l * D_ * D3, Wqkvp[l], D_, D3, DPAD, D3);
    packs(t_Wo   + (size_t)l * D_ * D_, Wop[l],   D_, D_, DPAD, D_);
    packs(t_W1   + (size_t)l * D_ * DFF, W1p[l],  D_, DFF, DPAD, DFF);
    packs(t_W2   + (size_t)l * DFF * D_, W2p[l],  DFF, D_, DFF, D_);
  }
  packs(mlp_W1, mW1p, DFIN, DFIN, DFINP, DFIN);
  packs(mlp_W2, mW2p, DFIN, 2, DFINP, 16);
  static_emb_kernel<<<(B_ * DSEMB + 255) / 256, 256, 0, stream>>>(statics, static_W, static_b, semb);

  // ---- graph propagation (2 layers), all GEMMs on WMMA ----
  for (int l = 0; l < 2; ++l) {
    const u16* Ain = (l == 0) ? hn : h1;
    dim3 gg((DFEAT / 16 + 15) / 16, S_ / 32, B_);
    gemm_bf16<false, false, true><<<gg, 128, 0, stream>>>(Ain, pwp[l][0], pB[l][0], nullptr, gq,
        DFEAT, DFEAT, DFEAT, 0, DFEAT, gstride, 0, gstride);
    gemm_bf16<false, false, true><<<gg, 128, 0, stream>>>(Ain, pwp[l][1], pB[l][1], nullptr, gk,
        DFEAT, DFEAT, DFEAT, 0, DFEAT, gstride, 0, gstride);
    gemm_bf16<false, false, true><<<gg, 128, 0, stream>>>(Ain, pwp[l][2], pB[l][2], nullptr, gv,
        DFEAT, DFEAT, DFEAT, 0, DFEAT, gstride, 0, gstride);
    gemm_bf16<false, false, true><<<gg, 128, 0, stream>>>(Ain, pwp[l][3], pB[l][3], nullptr, gs,
        DFEAT, DFEAT, DFEAT, 0, DFEAT, gstride, 0, gstride);
    graph_scores<<<B_, 256, 0, stream>>>(gk, gq, sc);
    if (l == 0) {
      graph_alpha<<<dim3(S_, B_), 64, 0, stream>>>(sc, nullptr, a1);
      graph_out<false><<<dim3(S_, B_), 256, 0, stream>>>(a1, gv, gs, h1, nullptr);
    } else {
      graph_alpha<<<dim3(S_, B_), 64, 0, stream>>>(sc, a1, a2);
      graph_out<true><<<dim3(S_, B_), 256, 0, stream>>>(a2, gv, gs, nullptr, xt);
    }
  }
  pe_kernel<<<(T_ * B_ + 255) / 256, 256, 0, stream>>>(times, xt);
  {
    size_t tot = (size_t)T_ * B_ * DPAD;
    f32_to_bf16_pad<<<(unsigned)((tot + 255) / 256), 256, 0, stream>>>(xt, xtb, T_ * B_, D_, DPAD);
    zero_u16<<<(unsigned)((tot + 255) / 256), 256, 0, stream>>>(ao, tot);
    zero_u16<<<(B_ * DFINP + 255) / 256, 256, 0, stream>>>(hidden, (size_t)B_ * DFINP);
  }

  // ---- transformer encoder (2 layers) ----
  const int MROW = T_ * B_;                       // 32768
  for (int l = 0; l < 2; ++l) {
    dim3 gqkv((D3 / 16 + 15) / 16, MROW / 32, 1);
    gemm_bf16<false, false, true><<<gqkv, 128, 0, stream>>>(xtb, Wqkvp[l], t_bqkv + (size_t)l * D3,
        nullptr, tqkv, DPAD, D3, D3, 0, D3, 0, 0, 0);
    attention<<<dim3(T_, B_, HEADS), 256, 0, stream>>>(tqkv, lengths, ao);
    dim3 gwo((D_ / 16 + 15) / 16, MROW / 32, 1);
    gemm_bf16<false, true, false><<<gwo, 128, 0, stream>>>(ao, Wop[l], t_bo + (size_t)l * D_,
        tmp, nullptr, DPAD, D_, D_, D_, 0, 0, 0, 0);
    residual_ln<<<MROW, 256, 0, stream>>>(xt, tmp, ln1g + (size_t)l * D_, ln1b + (size_t)l * D_, xt, xtb);
    dim3 gf1((DFF / 16 + 15) / 16, MROW / 32, 1);
    gemm_bf16<true, false, true><<<gf1, 128, 0, stream>>>(xtb, W1p[l], t_b1 + (size_t)l * DFF,
        nullptr, ff1, DPAD, DFF, DFF, 0, DFF, 0, 0, 0);
    dim3 gf2((D_ / 16 + 15) / 16, MROW / 32, 1);
    gemm_bf16<false, true, false><<<gf2, 128, 0, stream>>>(ff1, W2p[l], t_b2 + (size_t)l * D_,
        tmp, nullptr, DFF, D_, D_, D_, 0, 0, 0, 0);
    residual_ln<<<MROW, 256, 0, stream>>>(xt, tmp, ln2g + (size_t)l * D_, ln2b + (size_t)l * D_, xt, xtb);
  }

  // ---- pool + MLP head ----
  pool_feat<<<B_, 256, 0, stream>>>(xt, lengths, semb, featb);
  {
    dim3 g1((DFIN / 16 + 15) / 16, B_ / 32, 1);
    gemm_bf16<true, false, true><<<g1, 128, 0, stream>>>(featb, mW1p, mlp_b1,
        nullptr, hidden, DFINP, DFIN, DFIN, 0, DFINP, 0, 0, 0);
    dim3 g2(1, B_ / 32, 1);
    gemm_bf16<false, true, false><<<g2, 128, 0, stream>>>(hidden, mW2p, mlp_b2,
        out, nullptr, DFINP, 16, 2, 2, 0, 0, 0, 0);
  }
}
